// Paired_CGCNN_separated_19009525252276
// MI455X (gfx1250) — compile-verified, matrix operand order checked by reading, not run on hardware
//
#include <hip/hip_runtime.h>

typedef __attribute__((ext_vector_type(16))) _Float16 v16h;
typedef __attribute__((ext_vector_type(8)))  _Float16 h8;
typedef __attribute__((ext_vector_type(8)))  float    v8f;

#define NATOMS 20000
#define MNBR   12
#define ORIG   92
#define NBRF   41
#define FDIM   64
#define HDIM   128
#define NCONV  3
#define NCRYS  400
#define EPSBN  1e-5f

__device__ __forceinline__ float softplusf(float x) {
    return (x > 0.0f) ? (x + log1pf(expf(-x))) : log1pf(expf(x));
}
__device__ __forceinline__ float sigmoidf(float x) {
    return 1.0f / (1.0f + expf(-x));
}

// ---------------------------------------------------------------------------
// Pre-pass: convert f32 A (MxK) to f16 padded (MxKPAD), zero fill k>=K.
// ---------------------------------------------------------------------------
__global__ void pad_a_f16(const float* __restrict__ A, _Float16* __restrict__ Ah,
                          int total, int K, int KPAD) {
    int t = blockIdx.x * blockDim.x + threadIdx.x;
    if (t >= total) return;
    int k = t % KPAD;
    int m = t / KPAD;
    Ah[t] = (_Float16)((k < K) ? A[(size_t)m * K + k] : 0.0f);
}

// Pre-pass: B (KxN row-major f32) -> Bt (128 x KPAD, n-major f16, zero padded)
__global__ void pad_b_f16(const float* __restrict__ B, _Float16* __restrict__ Bt,
                          int K, int N, int KPAD) {
    int t = blockIdx.x * blockDim.x + threadIdx.x;
    if (t >= 128 * KPAD) return;
    int n = t / KPAD, k = t % KPAD;
    Bt[t] = (_Float16)((n < N && k < K) ? B[(size_t)k * N + n] : 0.0f);
}

// ---------------------------------------------------------------------------
// GEMM: C(M x N) = A(M x KPAD, f16) @ Bt(128 x KPAD, n-major f16) [+bias][sp]
// M must be a multiple of 16 (true for 20000 / 240000 / 400).
// Block = 256 threads = 8 waves; block tile 16 x 128, wave w -> cols w*16..+15.
// Fragments loaded directly from global as 2x b128 each (ISA 7.12.2 layout).
// ---------------------------------------------------------------------------
template <int KPAD>
__global__ __launch_bounds__(256)
void gemm_wmma_f16(const _Float16* __restrict__ A, const _Float16* __restrict__ Bt,
                   const float* __restrict__ bias, float* __restrict__ C,
                   _Float16* __restrict__ Ch, int N, int ldc, int act) {
    const int wave = threadIdx.x >> 5;
    const int lane = threadIdx.x & 31;
    const int mrow = lane & 15;   // A row / B col within tile
    const int kh   = lane >> 4;   // K-half selector
    const int m0   = blockIdx.x * 16;

    const _Float16* ap = A  + (size_t)(m0 + mrow) * KPAD + kh * 8;
    const _Float16* bp = Bt + (size_t)((wave << 4) + mrow) * KPAD + kh * 8;

    v8f acc = {};
#pragma unroll
    for (int k0 = 0; k0 < KPAD; k0 += 32) {
        union { v16h v; h8 h[2]; } a, b;
        a.h[0] = *(const h8*)(ap + k0);
        a.h[1] = *(const h8*)(ap + k0 + 16);
        b.h[0] = *(const h8*)(bp + k0);
        b.h[1] = *(const h8*)(bp + k0 + 16);
        acc = __builtin_amdgcn_wmma_f32_16x16x32_f16(
            false, a.v, false, b.v, (short)0, acc, false, false);
    }

    const int ncol = (wave << 4) + mrow;
    if (ncol < N) {
        float bv = bias ? bias[ncol] : 0.0f;
#pragma unroll
        for (int r = 0; r < 8; ++r) {
            int gm = m0 + r + kh * 8;
            float v = acc[r] + bv;
            if (act) v = softplusf(v);
            C[(size_t)gm * ldc + ncol] = v;
            if (Ch) Ch[(size_t)gm * ldc + ncol] = (_Float16)v;
        }
    }
}

// ---------------------------------------------------------------------------
// Conv combine: z = bn1(y1[i] + y2[idx] + y3[i,m] + b); s += sig(filt)*sp(core)
// a' = softplus(a + bn2(s)); also writes f16 copy of a' for the next GEMM.
// 64 threads per atom, 4 atoms per block. Fallback: inline nbr_fea @ W3.
// ---------------------------------------------------------------------------
__global__ __launch_bounds__(256)
void conv_combine(const float* __restrict__ a_cur, const float* __restrict__ y1,
                  const float* __restrict__ y2, const float* __restrict__ y3,
                  const float* __restrict__ nbr_fea, const int* __restrict__ nbr_idx,
                  const float* __restrict__ W3, const float* __restrict__ convb,
                  const float* __restrict__ bn1, const float* __restrict__ bn2,
                  float* __restrict__ a_nxt, _Float16* __restrict__ a_nxt_h,
                  int nAtoms) {
    __shared__ float sW3[NBRF * 128];  // only used in fallback
    if (y3 == nullptr) {
        for (int t = threadIdx.x; t < NBRF * 128; t += blockDim.x) sW3[t] = W3[t];
        __syncthreads();
    }
    const int atom = blockIdx.x * 4 + (threadIdx.x >> 6);
    const int c    = threadIdx.x & 63;
    if (atom >= nAtoms) return;

    const int cf = c, cc = c + 64;
    float sc_f = bn1[cf] * rsqrtf(bn1[384 + cf] + EPSBN);
    float sh_f = bn1[128 + cf] - bn1[256 + cf] * sc_f + convb[cf] * sc_f;
    float sc_c = bn1[cc] * rsqrtf(bn1[384 + cc] + EPSBN);
    float sh_c = bn1[128 + cc] - bn1[256 + cc] * sc_c + convb[cc] * sc_c;

    const float y1f = y1[(size_t)atom * 128 + cf];
    const float y1c = y1[(size_t)atom * 128 + cc];

    float s = 0.0f;
#pragma unroll
    for (int m = 0; m < MNBR; ++m) {
        const int row = atom * MNBR + m;
        const int j   = nbr_idx[row];
        float zf = y1f + y2[(size_t)j * 128 + cf];
        float zc = y1c + y2[(size_t)j * 128 + cc];
        if (y3) {
            zf += y3[(size_t)row * 128 + cf];
            zc += y3[(size_t)row * 128 + cc];
        } else {
            const float* nf = nbr_fea + (size_t)row * NBRF;
            float df = 0.0f, dc = 0.0f;
            for (int k = 0; k < NBRF; ++k) {
                float x = nf[k];
                df = fmaf(x, sW3[k * 128 + cf], df);
                dc = fmaf(x, sW3[k * 128 + cc], dc);
            }
            zf += df; zc += dc;
        }
        zf = zf * sc_f + sh_f;
        zc = zc * sc_c + sh_c;
        s += sigmoidf(zf) * softplusf(zc);
    }
    float sc2 = bn2[c] * rsqrtf(bn2[192 + c] + EPSBN);
    float sh2 = bn2[64 + c] - bn2[128 + c] * sc2;
    float out = softplusf(a_cur[(size_t)atom * 64 + c] + s * sc2 + sh2);
    a_nxt[(size_t)atom * 64 + c]   = out;
    a_nxt_h[(size_t)atom * 64 + c] = (_Float16)out;
}

// ---------------------------------------------------------------------------
// Pooling
// ---------------------------------------------------------------------------
__global__ void zero_buf(float* p, int n) {
    int i = blockIdx.x * blockDim.x + threadIdx.x;
    if (i < n) p[i] = 0.0f;
}
__global__ void pool_accum(const float* __restrict__ a, const int* __restrict__ seg,
                           float* __restrict__ sums, float* __restrict__ cnt) {
    int t = blockIdx.x * blockDim.x + threadIdx.x;
    if (t >= NATOMS * FDIM) return;
    int i = t >> 6, c = t & 63;
    int s = seg[i];
    atomicAdd(&sums[s * FDIM + c], a[(size_t)i * FDIM + c]);
    if (c == 0) atomicAdd(&cnt[s], 1.0f);
}
__global__ void pool_div(float* __restrict__ sums, const float* __restrict__ cnt,
                         _Float16* __restrict__ pooled_h) {
    int t = blockIdx.x * blockDim.x + threadIdx.x;
    if (t >= NCRYS * FDIM) return;
    int r = t >> 6;
    float v = sums[t] / fmaxf(cnt[r], 1.0f);
    pooled_h[t] = (_Float16)v;
}

// ---------------------------------------------------------------------------
// Head: d = |A - B|; h = softplus(d @ ffW + ffb); out = h @ outW + outb
// ---------------------------------------------------------------------------
__global__ __launch_bounds__(128)
void head_kernel(const float* __restrict__ crysA, const float* __restrict__ crysB,
                 const float* __restrict__ ffW, const float* __restrict__ ffb,
                 const float* __restrict__ outW, const float* __restrict__ outb,
                 float* __restrict__ out) {
    __shared__ float sd[HDIM];
    __shared__ float sr[HDIM];
    const int r = blockIdx.x, c = threadIdx.x;
    sd[c] = fabsf(crysA[(size_t)r * HDIM + c] - crysB[(size_t)r * HDIM + c]);
    __syncthreads();
    float h = ffb[c];
    for (int k = 0; k < HDIM; ++k) h = fmaf(sd[k], ffW[k * HDIM + c], h);
    h = softplusf(h);
    sr[c] = h * outW[c];
    __syncthreads();
    for (int st = 64; st > 0; st >>= 1) {
        if (c < st) sr[c] += sr[c + st];
        __syncthreads();
    }
    if (c == 0) out[r] = sr[0] + outb[0];
}

// ---------------------------------------------------------------------------
// Host launcher
// ---------------------------------------------------------------------------
static inline size_t align_up(size_t x) { return (x + 255) & ~(size_t)255; }

extern "C" void kernel_launch(void* const* d_in, const int* in_sizes, int n_in,
                              void* d_out, int out_size, void* d_ws, size_t ws_size,
                              hipStream_t stream) {
    (void)in_sizes; (void)n_in; (void)out_size;
    char* ws = (char*)d_ws;

    size_t off = 0;
    auto take = [&](size_t bytes) { size_t o = off; off += align_up(bytes); return o; };
    const size_t o_crysA  = take((size_t)NCRYS * HDIM * 4);
    const size_t o_crysB  = take((size_t)NCRYS * HDIM * 4);
    const size_t o_a0     = take((size_t)NATOMS * FDIM * 4);
    const size_t o_a1     = take((size_t)NATOMS * FDIM * 4);
    const size_t o_ah     = take((size_t)NATOMS * FDIM * 2);        // a in f16
    const size_t o_atomh  = take((size_t)NATOMS * 96 * 2);          // atom_fea f16, K 92->96
    const size_t o_y1     = take((size_t)NATOMS * 128 * 4);
    const size_t o_y2     = take((size_t)NATOMS * 128 * 4);
    const size_t o_pool   = take((size_t)(NCRYS * FDIM + NCRYS) * 4);
    const size_t o_poolh  = take((size_t)NCRYS * FDIM * 2);
    const size_t o_bt0    = take((size_t)128 * 96 * 2);             // weight tiles (n-major)
    const size_t o_bt1    = take((size_t)128 * 96 * 2);
    const size_t o_bt2    = take((size_t)128 * 96 * 2);
    const size_t o_nbrh   = off;                                    // gated on use_y3
    const size_t o_y3     = o_nbrh + align_up((size_t)NATOMS * MNBR * 64 * 2);
    const size_t need_y3  = o_y3 + (size_t)NATOMS * MNBR * 128 * 4;
    const bool use_y3 = (need_y3 <= ws_size);

    float*     crys[2]  = { (float*)(ws + o_crysA), (float*)(ws + o_crysB) };
    _Float16*  a_h      = (_Float16*)(ws + o_ah);
    _Float16*  atom_h   = (_Float16*)(ws + o_atomh);
    float*     y1buf    = (float*)(ws + o_y1);
    float*     y2buf    = (float*)(ws + o_y2);
    float*     sums     = (float*)(ws + o_pool);
    float*     cnt      = sums + NCRYS * FDIM;
    _Float16*  pooled_h = (_Float16*)(ws + o_poolh);
    _Float16*  bt[3]    = { (_Float16*)(ws + o_bt0), (_Float16*)(ws + o_bt1),
                            (_Float16*)(ws + o_bt2) };
    _Float16*  nbr_h    = use_y3 ? (_Float16*)(ws + o_nbrh) : nullptr;
    float*     y3buf    = use_y3 ? (float*)(ws + o_y3) : nullptr;

    for (int s = 0; s < 2; ++s) {
        const int b = s * 12;
        const float* atom_fea = (const float*)d_in[b + 0];
        const float* nbr_fea  = (const float*)d_in[b + 1];
        const int*   nbr_idx  = (const int*)  d_in[b + 2];
        const int*   seg      = (const int*)  d_in[b + 3];
        const float* emb_W    = (const float*)d_in[b + 4];
        const float* emb_b    = (const float*)d_in[b + 5];
        const float* conv_W   = (const float*)d_in[b + 6];
        const float* conv_b   = (const float*)d_in[b + 7];
        const float* bn1      = (const float*)d_in[b + 8];
        const float* bn2      = (const float*)d_in[b + 9];
        const float* dense_W  = (const float*)d_in[b + 10];
        const float* dense_b  = (const float*)d_in[b + 11];

        float* acur = (float*)(ws + o_a0);
        float* anxt = (float*)(ws + o_a1);

        // --- pre-pass: pad/convert A operands ---
        pad_a_f16<<<(NATOMS * 96 + 255) / 256, 256, 0, stream>>>(
            atom_fea, atom_h, NATOMS * 96, ORIG, 96);
        if (use_y3) {
            pad_a_f16<<<(NATOMS * MNBR * 64 + 255) / 256, 256, 0, stream>>>(
                nbr_fea, nbr_h, NATOMS * MNBR * 64, NBRF, 64);
        }

        // --- embedding: a0 = atom_fea @ emb_W + emb_b (also emits f16 copy) ---
        pad_b_f16<<<(128 * 96 + 255) / 256, 256, 0, stream>>>(emb_W, bt[0], ORIG, FDIM, 96);
        gemm_wmma_f16<96><<<NATOMS / 16, 256, 0, stream>>>(
            atom_h, bt[0], emb_b, acur, a_h, FDIM, FDIM, 0);

        for (int i = 0; i < NCONV; ++i) {
            const float* Wi = conv_W + (size_t)i * (2 * FDIM + NBRF) * 128;
            pad_b_f16<<<(128 * 64 + 255) / 256, 256, 0, stream>>>(Wi, bt[0], FDIM, 128, 64);
            pad_b_f16<<<(128 * 64 + 255) / 256, 256, 0, stream>>>(Wi + 64 * 128, bt[1], FDIM, 128, 64);
            gemm_wmma_f16<64><<<NATOMS / 16, 256, 0, stream>>>(
                a_h, bt[0], nullptr, y1buf, nullptr, 128, 128, 0);
            gemm_wmma_f16<64><<<NATOMS / 16, 256, 0, stream>>>(
                a_h, bt[1], nullptr, y2buf, nullptr, 128, 128, 0);
            if (use_y3) {
                pad_b_f16<<<(128 * 64 + 255) / 256, 256, 0, stream>>>(
                    Wi + 128 * 128, bt[2], NBRF, 128, 64);
                gemm_wmma_f16<64><<<(NATOMS * MNBR) / 16, 256, 0, stream>>>(
                    nbr_h, bt[2], nullptr, y3buf, nullptr, 128, 128, 0);
            }
            conv_combine<<<(NATOMS + 3) / 4, 256, 0, stream>>>(
                acur, y1buf, y2buf, y3buf, nbr_fea, nbr_idx,
                Wi + 128 * 128, conv_b + (size_t)i * 128,
                bn1 + (size_t)i * 4 * 128, bn2 + (size_t)i * 4 * 64,
                anxt, a_h, NATOMS);
            float* t = acur; acur = anxt; anxt = t;
        }

        // --- mean pool per crystal ---
        const int npool = NCRYS * FDIM + NCRYS;
        zero_buf<<<(npool + 255) / 256, 256, 0, stream>>>(sums, npool);
        pool_accum<<<(NATOMS * FDIM + 255) / 256, 256, 0, stream>>>(acur, seg, sums, cnt);
        pool_div<<<(NCRYS * FDIM + 255) / 256, 256, 0, stream>>>(sums, cnt, pooled_h);

        // --- dense: crys = softplus(pooled @ dense_W + dense_b) ---
        pad_b_f16<<<(128 * 64 + 255) / 256, 256, 0, stream>>>(dense_W, bt[0], FDIM, HDIM, 64);
        gemm_wmma_f16<64><<<NCRYS / 16, 256, 0, stream>>>(
            pooled_h, bt[0], dense_b, crys[s], nullptr, HDIM, HDIM, 1);
    }

    const float* ffW  = (const float*)d_in[24];
    const float* ffb  = (const float*)d_in[25];
    const float* outW = (const float*)d_in[26];
    const float* outb = (const float*)d_in[27];
    head_kernel<<<NCRYS, HDIM, 0, stream>>>(crys[0], crys[1], ffW, ffb, outW, outb,
                                            (float*)d_out);
}
